// SoftDTWCuda_69758858822517
// MI455X (gfx1250) — matrix operand order repaired and checked by reading, not compiled, not run on hardware
//
#include <hip/hip_runtime.h>

// ---------------------------------------------------------------------------
// Fused soft-DTW discrepancy for MI455X (gfx1250, wave32, WMMA, 320KB LDS/WGP)
// One workgroup per (pair, batch): pair 0 = (x,y), 1 = (x,x), 2 = (y,y).
// Everything (normalized f16 sequences, cost tiles, DP frontier) lives in LDS.
// ---------------------------------------------------------------------------

#define SLEN 512
#define DDIM 128
#define TILE 16
#define NB   (SLEN / TILE)   // 32 blocks per dimension
#define ROWP 136             // f16 row stride (128 + 8 pad -> 272B, breaks bank alignment)
#define INF_ 100000000.0f

typedef _Float16 v16h __attribute__((ext_vector_type(16)));
typedef _Float16 v8h  __attribute__((ext_vector_type(8)));
typedef float    v8f  __attribute__((ext_vector_type(8)));

// LDS layout (bytes / floats)
#define AN_BYTES     (SLEN * ROWP * 2)        // 139264
#define BN_BYTES     (SLEN * ROWP * 2)        // 139264
#define COST_FLOATS  (NB * TILE * 17)         // 32 tiles of 16x17 f32 = 8704
#define ROWBUF_FLOATS 516                     // DP row frontier R[i][0..512]
#define COLBUF_FLOATS 516                     // DP col frontier
#define CORNER_FLOATS (2 * 33)                // parity-double-buffered corners
#define SMEM_BYTES (AN_BYTES + BN_BYTES + \
                    (COST_FLOATS + ROWBUF_FLOATS + COLBUF_FLOATS + CORNER_FLOATS) * 4)

__global__ void zero_out_kernel(float* out, int n) {
    int i = blockIdx.x * blockDim.x + threadIdx.x;
    if (i < n) out[i] = 0.0f;
}

extern "C" __global__ void __launch_bounds__(512, 1)
softdtw_fused_kernel(const float* __restrict__ x, const float* __restrict__ y,
                     const float* __restrict__ gamma_p, float* __restrict__ out)
{
    extern __shared__ char smem[];
    _Float16* AN = (_Float16*)smem;                         // [512][136] f16
    _Float16* BN = (_Float16*)(smem + AN_BYTES);            // [512][136] f16
    float* COST      = (float*)(smem + AN_BYTES + BN_BYTES);// [32][16][17] f32
    float* rowbuf    = COST + COST_FLOATS;                  // R[row_below][j]
    float* colbuf    = rowbuf + ROWBUF_FLOATS;              // R[i][col_right]
    float* cornerbuf = colbuf + COLBUF_FLOATS;              // [2][33]

    const int pair = blockIdx.x;        // 0: xy, 1: xx, 2: yy
    const int b    = blockIdx.y;        // batch
    const int tid  = threadIdx.x;       // 0..511 (16 wave32 waves)
    const int lane = tid & 31;
    const int wv   = tid >> 5;          // wave id 0..15
    const int lmod = lane & 15;
    const int nhalf = lane >> 4;        // 0 for lanes 0-15, 1 for lanes 16-31

    const float g    = fmaxf(fabsf(gamma_p[0]), 1e-4f);
    const float invg = 1.0f / g;

    const float* asrc = (pair == 2) ? y : x;
    const float* bsrc = (pair == 1) ? x : y;

    // ---- Phase 0: row-normalize (cosine) into LDS as f16, one row per thread
    {
        const float4* ar = (const float4*)(asrc + ((size_t)b * SLEN + tid) * DDIM);
        const float4* br = (const float4*)(bsrc + ((size_t)b * SLEN + tid) * DDIM);
        float ssa = 0.0f, ssb = 0.0f;
        for (int k = 0; k < DDIM / 4; ++k) {
            float4 va = ar[k];
            ssa += va.x * va.x + va.y * va.y + va.z * va.z + va.w * va.w;
            float4 vb = br[k];
            ssb += vb.x * vb.x + vb.y * vb.y + vb.z * vb.z + vb.w * vb.w;
        }
        const float sa = 1.0f / fmaxf(sqrtf(ssa), 1e-8f);
        const float sb = 1.0f / fmaxf(sqrtf(ssb), 1e-8f);
        _Float16* anr = AN + (size_t)tid * ROWP;
        _Float16* bnr = BN + (size_t)tid * ROWP;
        for (int k = 0; k < DDIM / 4; ++k) {
            float4 va = ar[k];
            anr[4 * k + 0] = (_Float16)(va.x * sa);
            anr[4 * k + 1] = (_Float16)(va.y * sa);
            anr[4 * k + 2] = (_Float16)(va.z * sa);
            anr[4 * k + 3] = (_Float16)(va.w * sa);
            float4 vb = br[k];
            bnr[4 * k + 0] = (_Float16)(vb.x * sb);
            bnr[4 * k + 1] = (_Float16)(vb.y * sb);
            bnr[4 * k + 2] = (_Float16)(vb.z * sb);
            bnr[4 * k + 3] = (_Float16)(vb.w * sb);
        }
    }
    // DP frontier init: R[0][0]=0, R[0][j>0]=R[i>0][0]=INF, corners INF except [0][0]=0
    for (int i = tid; i < ROWBUF_FLOATS;  i += 512) rowbuf[i]    = (i == 0) ? 0.0f : INF_;
    for (int i = tid; i < COLBUF_FLOATS;  i += 512) colbuf[i]    = (i == 0) ? 0.0f : INF_;
    for (int i = tid; i < CORNER_FLOATS;  i += 512) cornerbuf[i] = (i == 0) ? 0.0f : INF_;
    __syncthreads();

    // ---- Phase 1: blocked anti-diagonal wavefront over 16x16 tiles ----------
    for (int t = 0; t < 2 * NB - 1; ++t) {
        const int bjlo = (t > NB - 1) ? (t - (NB - 1)) : 0;
        const int bjhi = (t < NB - 1) ? t : (NB - 1);

        // pre-read corners (written 2 diagonals ago, same parity slot)
        float creg[2];
        int nblk = 0;
        for (int bj = bjlo + wv; bj <= bjhi; bj += 16)
            creg[nblk++] = cornerbuf[(t & 1) * 33 + bj];
        __syncthreads();   // all corner reads before this diagonal's corner writes

        nblk = 0;
        for (int bj = bjlo + wv; bj <= bjhi; bj += 16) {
            const int bi = t - bj;
            const float corner = creg[nblk++];

            // --- cost tile via WMMA: cost = 1 - AN[bi-rows] . BN[bj-rows]^T ---
            v8f acc = {};
            const _Float16* arow = AN + (size_t)(bi * TILE + lmod) * ROWP;
            const _Float16* brow = BN + (size_t)(bj * TILE + lmod) * ROWP;
            #pragma unroll
            for (int k0 = 0; k0 < DDIM; k0 += 32) {
                // A 16x32 f16: lane half picks K {0..7,16..23} vs {8..15,24..31}
                v8h a0 = *(const v8h*)(arow + k0 + nhalf * 8);
                v8h a1 = *(const v8h*)(arow + k0 + 16 + nhalf * 8);
                // B 32x16 f16: lane = column, same K split as A {0..7,16..23} vs {8..15,24..31}
                v8h b0 = *(const v8h*)(brow + k0 + nhalf * 8);
                v8h b1 = *(const v8h*)(brow + k0 + 16 + nhalf * 8);
                v16h af, bf;
                #pragma unroll
                for (int i = 0; i < 8; ++i) {
                    af[i] = a0[i]; af[i + 8] = a1[i];
                    bf[i] = b0[i]; bf[i + 8] = b1[i];
                }
                acc = __builtin_amdgcn_wmma_f32_16x16x32_f16(
                    false, af, false, bf, (short)0, acc, false, false);
            }
            float* ct = COST + bj * (TILE * 17);
            #pragma unroll
            for (int gg = 0; gg < 8; ++gg) {
                const int m = gg + 8 * nhalf;           // C/D layout: M = g + 8*half
                ct[m * 17 + lmod] = 1.0f - acc[gg];
            }

            // --- in-tile DP wavefront: lane n owns column n (lanes 0..15) ---
            const int rb1 = bi * TILE + 1;              // first interior DP row
            const int cb1 = bj * TILE + 1;              // first interior DP col
            float cur = INF_, prev = INF_;
            float topv = 0.0f, topl = 0.0f;
            if (lane < 16) {
                topv = rowbuf[cb1 + lane];
                topl = (lane == 0) ? corner : rowbuf[cb1 + lane - 1];
            }
            for (int d = 0; d < 2 * TILE - 1; ++d) {
                const float lf_c = __shfl_up(cur, 1);   // left neighbor, 1 step old
                const float dg_p = __shfl_up(prev, 1);  // diag neighbor, 2 steps old
                if (lane < 16) {
                    const int m = d - lane;
                    if (m >= 0 && m < TILE) {
                        const float up = (m == 0) ? topv : cur;
                        float lf, dg;
                        if (lane == 0) {
                            lf = colbuf[rb1 + m];
                            dg = (m == 0) ? corner : colbuf[rb1 + m - 1];
                        } else {
                            lf = lf_c;
                            dg = (m == 0) ? topl : dg_p;
                        }
                        const float c  = ct[m * 17 + lane];
                        const float mn = fminf(up, fminf(dg, lf));
                        const float s  = __expf((mn - up) * invg)
                                       + __expf((mn - dg) * invg)
                                       + __expf((mn - lf) * invg);
                        const float nv = c + mn - g * __logf(s);
                        prev = cur;
                        cur  = nv;
                        if (lane == 15) colbuf[rb1 + m] = nv;  // export right column
                    }
                }
            }
            if (lane < 16) rowbuf[cb1 + lane] = cur;            // export bottom row
            if (lane == 15) cornerbuf[(t & 1) * 33 + bj + 1] = cur; // for (bi+1,bj+1)
        }
        __syncthreads();   // frontier for diagonal t complete
    }

    // ---- Phase 2: combine dist_xy - 0.5*(dist_xx + dist_yy) ----------------
    if (tid == 0) {
        const float dist  = rowbuf[SLEN];                       // R[512][512]
        const float coeff = (pair == 0) ? 1.0f : -0.5f;
        atomicAdd(&out[b], coeff * dist);
    }
}

extern "C" void kernel_launch(void* const* d_in, const int* in_sizes, int n_in,
                              void* d_out, int out_size, void* d_ws, size_t ws_size,
                              hipStream_t stream) {
    (void)in_sizes; (void)n_in; (void)d_ws; (void)ws_size;
    const float* x     = (const float*)d_in[0];
    const float* y     = (const float*)d_in[1];
    const float* gamma = (const float*)d_in[2];
    float* out = (float*)d_out;

    zero_out_kernel<<<1, 64, 0, stream>>>(out, out_size);

    dim3 grid(3, 32, 1);   // (pair, batch)
    dim3 block(512, 1, 1); // 16 wave32 waves
    softdtw_fused_kernel<<<grid, block, SMEM_BYTES, stream>>>(x, y, gamma, out);
}
